// LMPTLayer_77713138254282
// MI455X (gfx1250) — matrix operand order repaired and checked by reading, not compiled, not run on hardware
//
#include <hip/hip_runtime.h>
#include <hip/hip_bf16.h>

// ---------------------------------------------------------------------------
// Fused LMPT graph-transformer for MI455X (gfx1250, wave32, WMMA bf16).
//   T=3, P=3, B=16384, D=256, H=8, DH=32, NL=2, FF=1024, seq_len=3.
// One workgroup (256 thr = 8 waves) processes 16 sequences (48 rows) fully in
// LDS; weights are pre-converted to bf16 [out][in] in d_ws (L2-resident).
// Each wave owns 16-column blocks and computes all 3 row-tiles per block so
// B fragments (global/L2) are loaded once and reused for 3 WMMAs.
// ---------------------------------------------------------------------------

#define B_   16384
#define D_   256
#define FF_  1024
#define NL_  2
#define SEQW 16            // sequences per workgroup
#define RWS  48            // rows per workgroup (SEQW*3)
#define NWAVE 8
#define LDX  260           // float stride (residual / final-LN scratch)
#define LDA  264           // ushort stride (LN out / FFN hidden)
#define LDQ  776           // ushort stride (QKV: q|k|v = 3*256 + pad)

typedef __attribute__((ext_vector_type(8)))  unsigned short v8u16;
typedef __attribute__((ext_vector_type(16))) __bf16         v16bf;
typedef __attribute__((ext_vector_type(8)))  float          v8f;

__device__ __forceinline__ unsigned short f2bf(float f) {
  __bf16 h = (__bf16)f;                       // native v_cvt_*bf16_f32 on gfx1250
  return __builtin_bit_cast(unsigned short, h);
}
__device__ __forceinline__ float bf2f(unsigned short s) {
  return __uint_as_float(((unsigned int)s) << 16);
}

// A-matrix fragment (16-bit 16x32): lane half h holds contiguous K [16h,16h+16)
__device__ __forceinline__ v16bf load_frag_a(const unsigned short* p) {
  union { unsigned short s[16]; v16bf v; } u;
  v8u16 lo = *(const v8u16*)p;
  v8u16 hi = *(const v8u16*)(p + 8);
#pragma unroll
  for (int i = 0; i < 8; ++i) { u.s[i] = lo[i]; u.s[8 + i] = hi[i]; }
  return u.v;
}
// B-matrix fragment (16-bit 32x16): lane = column, lane half h holds contiguous K [16h,16h+16)
__device__ __forceinline__ v16bf load_frag_b(const unsigned short* p) {
  union { unsigned short s[16]; v16bf v; } u;
  v8u16 lo = *(const v8u16*)p;
  v8u16 hi = *(const v8u16*)(p + 8);
#pragma unroll
  for (int i = 0; i < 8; ++i) { u.s[i] = lo[i]; u.s[8 + i] = hi[i]; }
  return u.v;
}

// 3 row-tiles (48 rows) x one 16-col block, K = 256.
// A row-major in LDS (row 0 at aBase); B bf16 [n][k] in global (L2-hot).
// B fragments loaded once per K-step, reused for 3 WMMAs.
__device__ __forceinline__ void gemm3(const unsigned short* aBase, int lda,
                                      const unsigned short* bBase, int ldb,
                                      v8f acc[3], int lane) {
  const int m  = lane & 15;
  const int hf = lane >> 4;
  const unsigned short* ap = aBase + m * lda + 16 * hf;
  const unsigned short* bp = bBase + m * ldb + 16 * hf;
#pragma unroll
  for (int k = 0; k < 256; k += 32) {
    v16bf b = load_frag_b(bp + k);
#pragma unroll
    for (int mt = 0; mt < 3; ++mt) {
      v16bf a = load_frag_a(ap + mt * 16 * lda + k);
      acc[mt] = __builtin_amdgcn_wmma_f32_16x16x32_bf16(false, a, false, b,
                                                        (short)0, acc[mt], false, false);
    }
  }
}

// Row-wise LayerNorm over D=256: one wave per row, 8 elems per lane.
template <bool TO_BF16>
__device__ __forceinline__ void ln_rows(const float* x, void* dstv, int ldd,
                                        const float* g, const float* b,
                                        int lane, int wave) {
#pragma unroll 1
  for (int r = wave; r < RWS; r += NWAVE) {
    const float* row = x + r * LDX;
    float s = 0.f, ss = 0.f;
#pragma unroll
    for (int j = 0; j < 8; ++j) { float v = row[lane + 32 * j]; s += v; ss += v * v; }
#pragma unroll
    for (int o = 16; o > 0; o >>= 1) {
      s  += __shfl_xor(s,  o, 32);
      ss += __shfl_xor(ss, o, 32);
    }
    float mean = s * (1.f / 256.f);
    float var  = ss * (1.f / 256.f) - mean * mean;
    float rs   = rsqrtf(var + 1e-5f);
#pragma unroll
    for (int j = 0; j < 8; ++j) {
      int d = lane + 32 * j;
      float o_ = (row[d] - mean) * rs * g[d] + b[d];
      if (TO_BF16) ((unsigned short*)dstv)[r * ldd + d] = f2bf(o_);
      else         ((float*)dstv)[r * ldd + d] = o_;
    }
  }
}

struct Params {
  const float* term; const float* pred;
  const unsigned short* wq; const unsigned short* wk;
  const unsigned short* wv; const unsigned short* wo;
  const unsigned short* w1; const unsigned short* w2;
  const float* bq; const float* bk; const float* bv; const float* bo;
  const float* ln1g; const float* ln1b; const float* ln2g; const float* ln2b;
  const float* b1;   const float* b2;   const float* lnfg; const float* lnfb;
  float* out;
};

__global__ __launch_bounds__(256, 2) void lmpt_fused(Params p) {
  __shared__ alignas(16) float          s_xres[RWS * LDX];  //  49,920 B residual (fp32)
  __shared__ alignas(16) unsigned short s_xln [RWS * LDA];  //  25,344 B LN out / attn out (bf16)
  __shared__ alignas(16) unsigned short s_qkv [RWS * LDQ];  //  74,496 B QKV | FFN hidden | final scratch

  const int tid  = threadIdx.x;
  const int lane = tid & 31;
  const int wave = tid >> 5;
  const int s0   = blockIdx.x * SEQW;   // first sequence id for this WG
  const int nh   = lane & 15;           // C-tile column within 16
  const int hh4  = lane >> 4;           // C-tile row-half

  // ---- stage 0: NBP message construction -> residual stream --------------
  const int ATAB[9] = {1, 2, 0,  0, 2, 1,  1, 0, 2};
  const int PTAB[9] = {0, 2, 0,  0, 1, 0,  1, 2, 0};
  const int STAB[9] = {-1, -1, 0,  1, -1, 0,  1, 1, 0};
#pragma unroll 1
  for (int i = tid; i < RWS * D_; i += 256) {
    int lr = i >> 8, d = i & 255;
    int sl = lr / 3, tk = lr - 3 * sl;
    int n  = s0 + sl;
    int tm = n >> 14, b = n & (B_ - 1);
    int idx = tm * 3 + tk;
    float v = p.term[((size_t)ATAB[idx] * B_ + b) * D_ + d];
    int sg = STAB[idx];
    if (sg) v += (float)sg * p.pred[((size_t)PTAB[idx] * B_ + b) * D_ + d];
    s_xres[lr * LDX + d] = v;
  }
  __syncthreads();

#pragma unroll 1
  for (int l = 0; l < NL_; ++l) {
    const unsigned short* wq = p.wq + l * (D_ * D_);
    const unsigned short* wk = p.wk + l * (D_ * D_);
    const unsigned short* wv = p.wv + l * (D_ * D_);
    const unsigned short* wo = p.wo + l * (D_ * D_);
    const unsigned short* w1 = p.w1 + l * (D_ * FF_);
    const unsigned short* w2 = p.w2 + l * (D_ * FF_);
    const float* bq = p.bq + l * D_;
    const float* bk = p.bk + l * D_;
    const float* bv = p.bv + l * D_;
    const float* bo = p.bo + l * D_;
    const float* b1 = p.b1 + l * FF_;
    const float* b2 = p.b2 + l * D_;

    // ---- LN1 ------------------------------------------------------------
    ln_rows<true>(s_xres, s_xln, LDA, p.ln1g + l * D_, p.ln1b + l * D_, lane, wave);
    __syncthreads();

    // ---- QKV projections: wave owns column blocks, 3 row-tiles each ------
#pragma unroll
    for (int g3 = 0; g3 < 3; ++g3) {
      const unsigned short* wt   = (g3 == 0) ? wq : ((g3 == 1) ? wk : wv);
      const float*          bias = (g3 == 0) ? bq : ((g3 == 1) ? bk : bv);
#pragma unroll 1
      for (int nt = wave; nt < 16; nt += NWAVE) {
        float bb = bias[nt * 16 + nh];
        v8f acc[3];
#pragma unroll
        for (int mt = 0; mt < 3; ++mt)
#pragma unroll
          for (int v = 0; v < 8; ++v) acc[mt][v] = bb;
        gemm3(s_xln, LDA, wt + nt * 16 * D_, D_, acc, lane);
        unsigned short* dst = s_qkv + 8 * hh4 * LDQ + g3 * D_ + nt * 16 + nh;
#pragma unroll
        for (int mt = 0; mt < 3; ++mt)
#pragma unroll
          for (int v = 0; v < 8; ++v)
            dst[(mt * 16 + v) * LDQ] = f2bf(acc[mt][v]);
      }
    }
    __syncthreads();

    // ---- attention: 3x3 softmax per (seq, head), writes s_xln ------------
#pragma unroll 1
    for (int task = tid; task < SEQW * 8 * 3; task += 256) {
      int sidx = task / 24; int rem = task - sidx * 24;
      int hd = rem / 3, qi = rem % 3;
      int r0 = sidx * 3;
      const unsigned short* qrow = s_qkv + (r0 + qi) * LDQ + hd * 32;
      float qf[32];
#pragma unroll
      for (int d = 0; d < 32; ++d) qf[d] = bf2f(qrow[d]);
      float sc[3];
#pragma unroll
      for (int ki = 0; ki < 3; ++ki) {
        const unsigned short* krow = s_qkv + (r0 + ki) * LDQ + D_ + hd * 32;
        float a = 0.f;
#pragma unroll
        for (int d = 0; d < 32; ++d) a += qf[d] * bf2f(krow[d]);
        sc[ki] = a * 0.17677669529663687f;   // DH^-0.5
      }
      float mx = fmaxf(sc[0], fmaxf(sc[1], sc[2]));
      float e0 = __expf(sc[0] - mx), e1 = __expf(sc[1] - mx), e2 = __expf(sc[2] - mx);
      float inv = 1.f / (e0 + e1 + e2);
      e0 *= inv; e1 *= inv; e2 *= inv;
      const unsigned short* v0 = s_qkv + r0 * LDQ + 2 * D_ + hd * 32;
      unsigned short* orow = s_xln + (r0 + qi) * LDA + hd * 32;
#pragma unroll
      for (int d = 0; d < 32; ++d) {
        float ov = e0 * bf2f(v0[d]) + e1 * bf2f(v0[LDQ + d]) + e2 * bf2f(v0[2 * LDQ + d]);
        orow[d] = f2bf(ov);
      }
    }
    __syncthreads();

    // ---- O projection + residual add ------------------------------------
#pragma unroll 1
    for (int nt = wave; nt < 16; nt += NWAVE) {
      float bb = bo[nt * 16 + nh];
      v8f acc[3];
#pragma unroll
      for (int mt = 0; mt < 3; ++mt)
#pragma unroll
        for (int v = 0; v < 8; ++v) acc[mt][v] = bb;
      gemm3(s_xln, LDA, wo + nt * 16 * D_, D_, acc, lane);
      float* xr = s_xres + 8 * hh4 * LDX + nt * 16 + nh;
#pragma unroll
      for (int mt = 0; mt < 3; ++mt)
#pragma unroll
        for (int v = 0; v < 8; ++v)
          xr[(mt * 16 + v) * LDX] += acc[mt][v];
    }
    __syncthreads();

    // ---- LN2 ------------------------------------------------------------
    ln_rows<true>(s_xres, s_xln, LDA, p.ln2g + l * D_, p.ln2b + l * D_, lane, wave);
    __syncthreads();

    // ---- FFN, chunked over FF (4 x 256); hidden aliases dead QKV LDS -----
    unsigned short* h1 = s_qkv;
    v8f acc2[2][3];    // 2 column blocks per wave, 3 row-tiles each
#pragma unroll
    for (int j = 0; j < 2; ++j) {
      int nt = wave + j * NWAVE;
      float bb = b2[nt * 16 + nh];
#pragma unroll
      for (int mt = 0; mt < 3; ++mt)
#pragma unroll
        for (int v = 0; v < 8; ++v) acc2[j][mt][v] = bb;
    }
#pragma unroll 1
    for (int c = 0; c < 4; ++c) {
      // FFN1 chunk: relu(xln @ W1[:, c*256 : (c+1)*256]) -> h1 (bf16)
#pragma unroll 1
      for (int nt = wave; nt < 16; nt += NWAVE) {
        float bb = b1[c * 256 + nt * 16 + nh];
        v8f acc[3];
#pragma unroll
        for (int mt = 0; mt < 3; ++mt)
#pragma unroll
          for (int v = 0; v < 8; ++v) acc[mt][v] = bb;
        gemm3(s_xln, LDA, w1 + (c * 256 + nt * 16) * D_, D_, acc, lane);
        unsigned short* dst = h1 + 8 * hh4 * LDA + nt * 16 + nh;
#pragma unroll
        for (int mt = 0; mt < 3; ++mt)
#pragma unroll
          for (int v = 0; v < 8; ++v)
            dst[(mt * 16 + v) * LDA] = f2bf(fmaxf(acc[mt][v], 0.f));
      }
      __syncthreads();
      // FFN2 partial: acc2 += h1 @ W2[c*256 : (c+1)*256, :]
#pragma unroll
      for (int j = 0; j < 2; ++j) {
        int nt = wave + j * NWAVE;
        gemm3(h1, LDA, w2 + nt * 16 * FF_ + c * 256, FF_, acc2[j], lane);
      }
      __syncthreads();
    }
    // FFN residual epilogue
#pragma unroll
    for (int j = 0; j < 2; ++j) {
      int nt = wave + j * NWAVE;
      float* xr = s_xres + 8 * hh4 * LDX + nt * 16 + nh;
#pragma unroll
      for (int mt = 0; mt < 3; ++mt)
#pragma unroll
        for (int v = 0; v < 8; ++v)
          xr[(mt * 16 + v) * LDX] += acc2[j][mt][v];
    }
    __syncthreads();
  }

  // ---- final LN (fp32) + mean over 3 tokens -> global --------------------
  float* fbuf = (float*)s_qkv;
  ln_rows<false>(s_xres, fbuf, LDX, p.lnfg, p.lnfb, lane, wave);
  __syncthreads();
#pragma unroll 1
  for (int i = tid; i < SEQW * D_; i += 256) {
    int sidx = i >> 8, d = i & 255;
    const float* f0 = fbuf + (sidx * 3) * LDX + d;
    float m = (f0[0] + f0[LDX] + f0[2 * LDX]) * (1.f / 3.f);
    p.out[(size_t)(s0 + sidx) * D_ + d] = m;   // out flat index == n*D + d
  }
}

// ---------------------------------------------------------------------------
// Weight prep: fp32 [NL][R][C] -> bf16 [NL][C][R] in workspace (runs each call).
// ---------------------------------------------------------------------------
__global__ void cvt_transpose(const float* __restrict__ src,
                              unsigned short* __restrict__ dst, int R, int C) {
  int total = NL_ * R * C;
  for (int i = blockIdx.x * blockDim.x + threadIdx.x; i < total;
       i += gridDim.x * blockDim.x) {
    int l = i / (R * C); int rem = i - l * (R * C);
    int r = rem / C;     int c   = rem - r * C;
    dst[l * R * C + c * R + r] = f2bf(src[i]);
  }
}

extern "C" void kernel_launch(void* const* d_in, const int* in_sizes, int n_in,
                              void* d_out, int out_size, void* d_ws, size_t ws_size,
                              hipStream_t stream) {
  (void)in_sizes; (void)n_in; (void)out_size; (void)ws_size;
  const float* term = (const float*)d_in[0];
  const float* pred = (const float*)d_in[1];
  const float* Wq   = (const float*)d_in[2];
  const float* Wk   = (const float*)d_in[3];
  const float* Wv   = (const float*)d_in[4];
  const float* Wo   = (const float*)d_in[5];
  const float* bq   = (const float*)d_in[6];
  const float* bk   = (const float*)d_in[7];
  const float* bv   = (const float*)d_in[8];
  const float* bo   = (const float*)d_in[9];
  const float* ln1g = (const float*)d_in[10];
  const float* ln1b = (const float*)d_in[11];
  const float* ln2g = (const float*)d_in[12];
  const float* ln2b = (const float*)d_in[13];
  const float* W1   = (const float*)d_in[14];
  const float* b1   = (const float*)d_in[15];
  const float* W2   = (const float*)d_in[16];
  const float* b2   = (const float*)d_in[17];
  const float* lnfg = (const float*)d_in[18];
  const float* lnfb = (const float*)d_in[19];

  unsigned short* ws  = (unsigned short*)d_ws;
  unsigned short* wqT = ws;
  unsigned short* wkT = wqT + NL_ * D_ * D_;
  unsigned short* wvT = wkT + NL_ * D_ * D_;
  unsigned short* woT = wvT + NL_ * D_ * D_;
  unsigned short* w1T = woT + NL_ * D_ * D_;
  unsigned short* w2T = w1T + NL_ * D_ * FF_;   // total ws use: 3,145,728 B

  cvt_transpose<<<256, 256, 0, stream>>>(Wq, wqT, D_, D_);
  cvt_transpose<<<256, 256, 0, stream>>>(Wk, wkT, D_, D_);
  cvt_transpose<<<256, 256, 0, stream>>>(Wv, wvT, D_, D_);
  cvt_transpose<<<256, 256, 0, stream>>>(Wo, woT, D_, D_);
  cvt_transpose<<<512, 256, 0, stream>>>(W1, w1T, D_, FF_);
  cvt_transpose<<<512, 256, 0, stream>>>(W2, w2T, FF_, D_);

  Params p{term, pred, wqT, wkT, wvT, woT, w1T, w2T,
           bq, bk, bv, bo, ln1g, ln1b, ln2g, ln2b,
           b1, b2, lnfg, lnfb, (float*)d_out};

  const int nblocks = (3 * B_) / SEQW;  // 49152 sequences / 16 = 3072 WGs
  lmpt_fused<<<nblocks, 256, 0, stream>>>(p);
}
